// Graph_RGSN_6571299963188
// MI455X (gfx1250) — matrix operand order repaired and checked by reading, not compiled, not run on hardware
//
#include <hip/hip_runtime.h>
#include <hip/hip_bf16.h>

#define N_NODES   50000
#define N_EDGES   600000
#define N_RELS    8
#define ID_DIM    64
#define IN_DIM    128
#define HID       128
#define NLAYERS   3
#define N_GRAPHS  64
#define BN_EPS    1e-5f
#define OUT_COLS  (HID * NLAYERS)      // 384

#define PAD_CAP   (N_EDGES + N_RELS * 16)   // 600128, every relation segment 16-aligned
#define N_TILES   (PAD_CAP / 16)            // 37508
#define A_STRIDE  208                       // LDS row stride in halves (>=192, conflict-padded)

typedef _Float16 half_t;
typedef __attribute__((ext_vector_type(16))) _Float16 v16h;
typedef __attribute__((ext_vector_type(8)))  _Float16 v8h;
typedef __attribute__((ext_vector_type(8)))  float    v8f;

// ---------------- small utility kernels ----------------

__global__ void k_zero_f32(float* p, size_t n) {
    size_t i = (size_t)blockIdx.x * blockDim.x + threadIdx.x;
    size_t stride = (size_t)gridDim.x * blockDim.x;
    for (; i < n; i += stride) p[i] = 0.0f;
}

__global__ void k_fill_i32(int* p, int v, size_t n) {
    size_t i = (size_t)blockIdx.x * blockDim.x + threadIdx.x;
    if (i < n) p[i] = v;
}

__global__ void k_hist_zero(int* hist) {
    if (threadIdx.x < N_RELS) hist[threadIdx.x] = 0;
}

__global__ void k_hist(const int* __restrict__ etype, int* __restrict__ hist) {
    int i = blockIdx.x * blockDim.x + threadIdx.x;
    if (i < N_EDGES) atomicAdd(&hist[etype[i]], 1);
}

// single-thread exclusive scan with 16-alignment; also initializes scatter cursors
__global__ void k_scan(const int* __restrict__ hist, int* __restrict__ pad_start,
                       int* __restrict__ cursor) {
    if (blockIdx.x == 0 && threadIdx.x == 0) {
        int o = 0;
        for (int r = 0; r < N_RELS; ++r) {
            pad_start[r] = o;
            cursor[r]    = o;
            o += (hist[r] + 15) & ~15;
        }
        pad_start[N_RELS] = o;
    }
}

__global__ void k_scatter(const int* __restrict__ src, const int* __restrict__ nbr,
                          const int* __restrict__ etype, int* __restrict__ cursor,
                          int* __restrict__ ssrc, int* __restrict__ snbr) {
    int i = blockIdx.x * blockDim.x + threadIdx.x;
    if (i < N_EDGES) {
        int r = etype[i];
        int pos = atomicAdd(&cursor[r], 1);
        ssrc[pos] = src[i];
        snbr[pos] = nbr[i];
    }
}

__global__ void k_convert_x(const float* __restrict__ x, half_t* __restrict__ h0) {
    size_t i = (size_t)blockIdx.x * blockDim.x + threadIdx.x;
    if (i < (size_t)N_NODES * IN_DIM) h0[i] = (half_t)x[i];
}

// Pack W[r][din][128] (f32 row-major) into per-lane WMMA-B fragment order (f16):
// pack[(((r*nkc + kc)*8 + ntile)*32 + lane)*16 + t]
//   lane = nn + 16*(kk>=16), t = kk&15, element = W[r][kc*32+kk][ntile*16+nn]
__global__ void k_pack_w(const float* __restrict__ W, half_t* __restrict__ pack, int din) {
    int total = N_RELS * din * HID;
    int idx = blockIdx.x * blockDim.x + threadIdx.x;
    if (idx >= total) return;
    int t     = idx & 15;
    int lane  = (idx >> 4) & 31;
    int ntile = (idx >> 9) & 7;
    int rest  = idx >> 12;
    int nkc   = din >> 5;
    int kc    = rest % nkc;
    int r     = rest / nkc;
    int kk    = ((lane >> 4) << 4) + t;     // 0..31
    int nn    = lane & 15;
    pack[idx] = (half_t)W[((size_t)r * din + kc * 32 + kk) * HID + ntile * 16 + nn];
}

// ---------------- main RGCN tile-GEMM kernel (WMMA) ----------------
// block = 256 threads (8 waves). One block: 16 edges (single relation) x 128 outputs.
// wave w handles output columns [16w, 16w+16). K-loop of v_wmma_f32_16x16x32_f16.
__global__ __launch_bounds__(256) void k_rgcn_gemm(
    const half_t* __restrict__ h, int din,
    const half_t* __restrict__ wpack,
    const int* __restrict__ ssrc, const int* __restrict__ snbr,
    const int* __restrict__ pad_start,
    float* __restrict__ z)
{
    __shared__ __align__(16) half_t sA[16 * A_STRIDE];
    __shared__ int s_src[16];

    int base = blockIdx.x * 16;
    if (base >= pad_start[N_RELS]) return;          // uniform per block
    int r = 0;
    while (r < N_RELS - 1 && base >= pad_start[r + 1]) ++r;

    int tid = threadIdx.x;
    int uints_per_row = din >> 1;                   // din is even (128/192)
    // cooperative gather of 16 neighbor rows (f16) into LDS; padded edges -> zeros
    for (int i = tid; i < 16 * uints_per_row; i += 256) {
        int row = i / uints_per_row;
        int d   = i - row * uints_per_row;
        int node = snbr[base + row];
        unsigned int v = 0u;
        if (node >= 0) v = ((const unsigned int*)(h + (size_t)node * din))[d];
        ((unsigned int*)(sA + row * A_STRIDE))[d] = v;
    }
    if (tid < 16) s_src[tid] = ssrc[base + tid];
    __syncthreads();

    int wave  = tid >> 5;          // n-tile (0..7)
    int lane  = tid & 31;
    int arow  = lane & 15;         // A matrix row (edge within tile)
    int khalf = lane >> 4;         // 0: K 0-15 ; 1: K 16-31

    v8f acc = {};
    int nkc = din >> 5;
    const half_t* wbase = wpack + ((size_t)r * nkc * 8 + wave) * 512 + lane * 16;

    for (int kc = 0; kc < nkc; ++kc) {
        const half_t* ap = sA + arow * A_STRIDE + kc * 32 + khalf * 16;
        v8h a0 = *(const v8h*)ap;          // K base+0..7   (or 16..23)
        v8h a1 = *(const v8h*)(ap + 8);    // K base+8..15  (or 24..31)
        v16h a = __builtin_shufflevector(a0, a1,
                 0, 1, 2, 3, 4, 5, 6, 7, 8, 9, 10, 11, 12, 13, 14, 15);
        v16h b = *(const v16h*)(wbase + (size_t)kc * 8 * 512);
#if defined(__HIP_DEVICE_COMPILE__)
        acc = __builtin_amdgcn_wmma_f32_16x16x32_f16(
                  false, a, false, b, (short)0, acc, false, false);
#endif
    }

    // per-edge ReLU then scatter-add into z[src] (atomics resolve in L2)
    int col = lane & 15;
#pragma unroll
    for (int j = 0; j < 8; ++j) {
        int erow = j + 8 * khalf;          // C layout: VGPR j, lanes16-31 -> M=j+8
        int s = s_src[erow];
        float v = acc[j];
        v = v > 0.0f ? v : 0.0f;
        if (s >= 0) atomicAdd(&z[(size_t)s * HID + wave * 16 + col], v);
    }
}

// ---------------- batch-norm + epilogue ----------------

__global__ __launch_bounds__(256) void k_bn_partial(const float* __restrict__ z,
                                                    float* __restrict__ sums) {
    int tid = threadIdx.x;
    int c   = tid & 127;
    int sub = tid >> 7;                 // 2 nodes per block iteration, coalesced
    float s = 0.0f, q = 0.0f;
    for (int n = blockIdx.x * 2 + sub; n < N_NODES; n += gridDim.x * 2) {
        float v = z[(size_t)n * HID + c];
        s += v; q += v * v;
    }
    atomicAdd(&sums[c], s);
    atomicAdd(&sums[128 + c], q);
}

__global__ void k_bn_finalize(const float* __restrict__ sums, float* __restrict__ meanvar) {
    int c = threadIdx.x;                // 128
    float m = sums[c] / (float)N_NODES;
    float v = sums[128 + c] / (float)N_NODES - m * m;   // biased variance
    meanvar[c]       = m;
    meanvar[128 + c] = rsqrtf(v + BN_EPS);
}

// normalize + ReLU, write z_cat slice, atomic segment-sum into g_cat,
// and build next layer's f16 input [x_id(64) | z(128)]
__global__ __launch_bounds__(128) void k_bn_apply(
    const float* __restrict__ z, const float* __restrict__ meanvar,
    const float* __restrict__ gamma, const float* __restrict__ beta,
    const float* __restrict__ x, const int* __restrict__ batch,
    float* __restrict__ out_z, float* __restrict__ out_g,
    half_t* __restrict__ h_next, int layer)
{
    int n = blockIdx.x;
    int c = threadIdx.x;
    float v = (z[(size_t)n * HID + c] - meanvar[c]) * meanvar[128 + c]
              * gamma[layer * HID + c] + beta[layer * HID + c];
    v = v > 0.0f ? v : 0.0f;
    out_z[(size_t)n * OUT_COLS + layer * HID + c] = v;
    atomicAdd(&out_g[(size_t)batch[n] * OUT_COLS + layer * HID + c], v);
    if (h_next) {
        h_next[(size_t)n * (ID_DIM + HID) + ID_DIM + c] = (half_t)v;
        if (c < ID_DIM)
            h_next[(size_t)n * (ID_DIM + HID) + c] = (half_t)x[(size_t)n * IN_DIM + c];
    }
}

// ---------------- launcher ----------------

extern "C" void kernel_launch(void* const* d_in, const int* in_sizes, int n_in,
                              void* d_out, int out_size, void* d_ws, size_t ws_size,
                              hipStream_t stream) {
    const float* x     = (const float*)d_in[0];
    const float* W0    = (const float*)d_in[1];
    const float* W1    = (const float*)d_in[2];
    const float* W2    = (const float*)d_in[3];
    const float* gamma = (const float*)d_in[4];
    const float* beta  = (const float*)d_in[5];
    const int*   eidx  = (const int*)d_in[6];   // [2][N_EDGES]: row0=src, row1=nbr
    const int*   etype = (const int*)d_in[7];
    const int*   batch = (const int*)d_in[8];

    float* out_z = (float*)d_out;                              // [N_NODES, 384]
    float* out_g = out_z + (size_t)N_NODES * OUT_COLS;         // [64, 384]

    // workspace carve-up (256B aligned)
    char* wsb = (char*)d_ws;
    size_t off = 0;
    auto take = [&](size_t bytes) -> void* {
        void* p = wsb + off;
        off = (off + bytes + 255) & ~(size_t)255;
        return p;
    };
    int*    hist      = (int*)take(N_RELS * 4);
    int*    cursor    = (int*)take(N_RELS * 4);
    int*    pad_start = (int*)take((N_RELS + 1) * 4);
    int*    ssrc      = (int*)take((size_t)PAD_CAP * 4);
    int*    snbr      = (int*)take((size_t)PAD_CAP * 4);
    half_t* wp0       = (half_t*)take((size_t)N_RELS * 128 * HID * 2);
    half_t* wp1       = (half_t*)take((size_t)N_RELS * 192 * HID * 2);
    half_t* wp2       = (half_t*)take((size_t)N_RELS * 192 * HID * 2);
    half_t* hbuf      = (half_t*)take((size_t)N_NODES * (ID_DIM + HID) * 2);
    float*  zbuf      = (float*)take((size_t)N_NODES * HID * 4);
    float*  sums      = (float*)take(256 * 4);
    float*  meanvar   = (float*)take(256 * 4);
    (void)ws_size; (void)n_in; (void)in_sizes; (void)out_size;

    const int* esrc = eidx;
    const int* enbr = eidx + N_EDGES;

    // 1) counting-sort edges by relation into 16-aligned segments
    k_hist_zero<<<1, N_RELS, 0, stream>>>(hist);
    k_hist<<<(N_EDGES + 255) / 256, 256, 0, stream>>>(etype, hist);
    k_scan<<<1, 1, 0, stream>>>(hist, pad_start, cursor);
    k_fill_i32<<<(PAD_CAP + 255) / 256, 256, 0, stream>>>(ssrc, -1, PAD_CAP);
    k_fill_i32<<<(PAD_CAP + 255) / 256, 256, 0, stream>>>(snbr, -1, PAD_CAP);
    k_scatter<<<(N_EDGES + 255) / 256, 256, 0, stream>>>(esrc, enbr, etype, cursor, ssrc, snbr);

    // 2) precision / layout prep
    k_convert_x<<<((size_t)N_NODES * IN_DIM + 255) / 256, 256, 0, stream>>>(x, hbuf);
    k_pack_w<<<(N_RELS * 128 * HID + 255) / 256, 256, 0, stream>>>(W0, wp0, 128);
    k_pack_w<<<(N_RELS * 192 * HID + 255) / 256, 256, 0, stream>>>(W1, wp1, 192);
    k_pack_w<<<(N_RELS * 192 * HID + 255) / 256, 256, 0, stream>>>(W2, wp2, 192);

    // 3) zero the segment-sum output region (atomically accumulated)
    k_zero_f32<<<96, 256, 0, stream>>>(out_g, (size_t)N_GRAPHS * OUT_COLS);

    // 4) three RGCN layers
    half_t* wps[NLAYERS] = { wp0, wp1, wp2 };
    for (int layer = 0; layer < NLAYERS; ++layer) {
        int din = (layer == 0) ? IN_DIM : (ID_DIM + HID);
        k_zero_f32<<<25000, 256, 0, stream>>>(zbuf, (size_t)N_NODES * HID);
        k_rgcn_gemm<<<N_TILES, 256, 0, stream>>>(hbuf, din, wps[layer],
                                                 ssrc, snbr, pad_start, zbuf);
        k_zero_f32<<<1, 256, 0, stream>>>(sums, 256);
        k_bn_partial<<<200, 256, 0, stream>>>(zbuf, sums);
        k_bn_finalize<<<1, 128, 0, stream>>>(sums, meanvar);
        k_bn_apply<<<N_NODES, 128, 0, stream>>>(zbuf, meanvar, gamma, beta, x, batch,
                                                out_z, out_g,
                                                (layer < NLAYERS - 1) ? hbuf : (half_t*)nullptr,
                                                layer);
    }
}
